// PatchEmbedding_4690104287386
// MI455X (gfx1250) — matrix operand order repaired and checked by reading, not compiled, not run on hardware
//
#include <hip/hip_runtime.h>

typedef __attribute__((ext_vector_type(16))) _Float16 v16h;
typedef __attribute__((ext_vector_type(8)))  _Float16 v8h;
typedef __attribute__((ext_vector_type(8)))  float    v8f;

#define TRY_ASYNC 1   // 1: global_load_async_to_lds_b128 (ASYNCcnt); 0: load+ds_store fallback

// Problem constants
#define NPATCH   196            // 14*14 patches per image
#define MTOT     12544          // 64*196 rows
#define DIMD     384            // embed dim
#define DIMK     768            // 3*16*16
#define NT       24             // DIMD/16 N-tiles
#define KSTEPS   24             // DIMK/32 K-steps
#define ROWS_WG  64             // patch rows per workgroup
#define ROW_PAD  776            // 768 + 8 halves -> 1552B row stride (bank-conflict-free b128 frags)

#define AS_BYTES    (ROWS_WG * ROW_PAD * 2)   // 99328
#define HIST_OFF    AS_BYTES
#define HIST_BYTES  (ROWS_WG * 32 * 4)        // 8192
#define LN_OFF      (HIST_OFF + HIST_BYTES)
#define LN_BYTES    (ROWS_WG * 4 * 4)         // 1024
#define BBUF_OFF    (LN_OFF + LN_BYTES)       // 108544
#define BSTEP_BYTES (NT * 32 * 32)            // 24576 per K-step (24 tiles x 1KB)
#define SMEM_BYTES  (BBUF_OFF + 2 * BSTEP_BYTES)  // 157696 -> 2 WGs/WGP (320KB LDS)

// ---------------------------------------------------------------------------
// Pack proj_w (384x768 f32) into the LDS-image B layout, f16:
// per (ks, t): 1KB block = [512B: lo-halves (8 halves/lane, lane-stride 16B)]
//                          [512B: hi-halves]. Lane holds column n = t*16+(lane&15),
// K-range ks*32 + (lane<16 ? 0 : 16) .. +15 (contiguous k order, same as A).
// 576KB total -> L2-resident; async-copied 24KB/K-step into LDS ping-pong.
// ---------------------------------------------------------------------------
__global__ void pack_w_kernel(const float* __restrict__ W, _Float16* __restrict__ out) {
    int g = blockIdx.x * 256 + threadIdx.x;        // 24*24*32 = 18432 items
    if (g >= NT * KSTEPS * 32) return;
    int lane = g & 31;
    int ks   = (g >> 5) % KSTEPS;
    int t    = g / (32 * KSTEPS);
    int kh   = lane >> 4;
    int n    = t * 16 + (lane & 15);
    int kbase = ks * 32 + kh * 16;

    union { v8h v; _Float16 e[8]; } lo, hi;
#pragma unroll
    for (int j = 0; j < 8; ++j) {
        lo.e[j] = (_Float16)W[(size_t)n * DIMK + kbase + j];
        hi.e[j] = (_Float16)W[(size_t)n * DIMK + kbase + 8 + j];
    }
    _Float16* base = out + ((size_t)(ks * NT + t)) * 512;   // 1KB block (512 halves)
    *(v8h*)(base + lane * 8)       = lo.v;
    *(v8h*)(base + 256 + lane * 8) = hi.v;
}

// ---------------------------------------------------------------------------
// Issue the cooperative copy of one K-step's B block (24KB) into LDS.
// 48 chunks of 512B (32 lanes x 16B); wave w takes chunks w, w+8, ..., w+40.
// ---------------------------------------------------------------------------
__device__ __forceinline__ void bcopy_issue(const _Float16* __restrict__ wsB, int ks,
                                            char* ldsdst, int w, int lane) {
    const char* gsrc = (const char*)wsB + (size_t)ks * BSTEP_BYTES;
#pragma unroll
    for (int j = 0; j < 6; ++j) {
        int off = (w + j * 8) * 512 + lane * 16;
#if TRY_ASYNC
        unsigned long long ga = (unsigned long long)(uintptr_t)(gsrc + off);
        unsigned           la = (unsigned)(uintptr_t)(ldsdst + off);  // low 32 bits = LDS offset
        asm volatile("global_load_async_to_lds_b128 %0, %1, off"
                     :: "v"(la), "v"(ga) : "memory");
#else
        *(v8h*)(ldsdst + off) = *(const v8h*)(gsrc + off);
#endif
    }
}

// ---------------------------------------------------------------------------
// Fused: stage patches f32->f16 into LDS, per-patch entropy from LDS,
// WMMA GEMM (f16 in, f32 acc, B double-buffered in LDS), bias + LayerNorm.
// Workgroup: 256 threads = 8 waves; 64 patch rows; all 384 columns.
//   wave w: rows (w>>1)*16 .. +15, N-tiles t0=(w&1)*12 .. +11
// ---------------------------------------------------------------------------
__global__ void __launch_bounds__(256)
vit_embed_kernel(const float* __restrict__ img,
                 const float* __restrict__ bias,
                 const float* __restrict__ gamma,
                 const float* __restrict__ beta,
                 const _Float16* __restrict__ wsB,
                 float* __restrict__ out) {
    extern __shared__ char smem[];
    _Float16* As   = (_Float16*)smem;                 // [64][776] halves
    unsigned* Hist = (unsigned*)(smem + HIST_OFF);    // [64][32]
    float*    LN   = (float*)(smem + LN_OFF);         // [64][4]: sumL,sqL,sumR,sqR
    char*     Bbuf = smem + BBUF_OFF;                 // 2 x 24KB ping-pong

    const int tid  = threadIdx.x;
    const int w    = tid >> 5;
    const int lane = tid & 31;
    const int m0   = blockIdx.x * ROWS_WG;            // 196 blocks * 64 = 12544

    // Kick off async copy of K-step 0's B block; it overlaps the whole
    // staging + entropy phase below.
    bcopy_issue(wsB, 0, Bbuf, w, lane);

    // ---- zero histograms ----
    for (int i = tid; i < ROWS_WG * 32; i += 256) Hist[i] = 0u;

    // ---- stage A-tile: 64 rows x 768 K, f32 -> f16 ----
#pragma unroll 4
    for (int it = 0; it < 12; ++it) {
        int id  = it * 256 + tid;                     // 0..3071
        int py  = id & 15;
        int c   = (id >> 4) % 3;
        int row = id / 48;
        int gp  = m0 + row;
        int b   = gp / NPATCH, pp = gp % NPATCH;
        int gy  = pp / 14,     gx = pp % 14;
        const float* src = img + ((((size_t)b * 3 + c) * 224) + gy * 16 + py) * 224 + gx * 16;
        float4 f0 = ((const float4*)src)[0];
        float4 f1 = ((const float4*)src)[1];
        float4 f2 = ((const float4*)src)[2];
        float4 f3 = ((const float4*)src)[3];
        union { v8h v; _Float16 e[8]; } lo, hi;
        lo.e[0]=(_Float16)f0.x; lo.e[1]=(_Float16)f0.y; lo.e[2]=(_Float16)f0.z; lo.e[3]=(_Float16)f0.w;
        lo.e[4]=(_Float16)f1.x; lo.e[5]=(_Float16)f1.y; lo.e[6]=(_Float16)f1.z; lo.e[7]=(_Float16)f1.w;
        hi.e[0]=(_Float16)f2.x; hi.e[1]=(_Float16)f2.y; hi.e[2]=(_Float16)f2.z; hi.e[3]=(_Float16)f2.w;
        hi.e[4]=(_Float16)f3.x; hi.e[5]=(_Float16)f3.y; hi.e[6]=(_Float16)f3.z; hi.e[7]=(_Float16)f3.w;
        _Float16* dst = As + (size_t)row * ROW_PAD + c * 256 + py * 16;
        *(v8h*)dst       = lo.v;
        *(v8h*)(dst + 8) = hi.v;
    }
    __syncthreads();

    // ---- per-patch entropy from staged LDS (no extra HBM read) ----
    {
        int p = tid >> 2;
        int q = tid & 3;
        const _Float16* base = As + (size_t)p * ROW_PAD;
#pragma unroll
        for (int jg = 0; jg < 8; ++jg) {
            int pix = q * 64 + jg * 8;
            v8h c0 = *(const v8h*)(base + pix);
            v8h c1 = *(const v8h*)(base + 256 + pix);
            v8h c2 = *(const v8h*)(base + 512 + pix);
#pragma unroll
            for (int e = 0; e < 8; ++e) {
                float gray = ((float)c0[e] + (float)c1[e] + (float)c2[e]) * (1.0f / 3.0f);
                int bin = (int)(gray * 31.0f);
                bin = bin < 0 ? 0 : (bin > 31 ? 31 : bin);
                atomicAdd(&Hist[p * 32 + bin], 1u);
            }
        }
    }
    __syncthreads();

    if (tid < ROWS_WG) {                  // finalize entropy, 1 thread / patch
        float ent = 0.0f;
#pragma unroll
        for (int bin = 0; bin < 32; ++bin) {
            unsigned c = Hist[tid * 32 + bin];
            if (c) {
                float p = (float)c * (1.0f / 256.0f);
                ent -= p * __log2f(p + 1e-10f);
            }
        }
        out[(size_t)MTOT * DIMD + m0 + tid] = ent * 0.2f;   // / log2(32)
    }

    // ---- WMMA GEMM with B double-buffered through LDS ----
    const int kh  = lane >> 4;            // K-half select
    const int c16 = lane & 15;            // A-row / B-col within tile
    const int mrl = (w >> 1) * 16;        // LDS row base of this wave's M-tile
    const int t0  = (w & 1) * 12;         // first N-tile

    float bv[12];
#pragma unroll
    for (int i = 0; i < 12; ++i) bv[i] = bias[(t0 + i) * 16 + c16];

    v8f acc[12] = {};
    const _Float16* arow = As + (size_t)(mrl + c16) * ROW_PAD + kh * 16;

#pragma unroll 1
    for (int ks = 0; ks < KSTEPS; ++ks) {
#if TRY_ASYNC
        asm volatile("s_wait_asynccnt 0x0" ::: "memory");   // buf[ks&1] copies done
#endif
        __syncthreads();   // all waves: copies visible AND prev-step readers done
        if (ks + 1 < KSTEPS)
            bcopy_issue(wsB, ks + 1, Bbuf + ((ks + 1) & 1) * BSTEP_BYTES, w, lane);

        union { v16h v; v8h h[2]; } a;
        const _Float16* ap = arow + ks * 32;
        a.h[0] = *(const v8h*)ap;            // K k0+kh*16 .. +7
        a.h[1] = *(const v8h*)(ap + 8);      // K k0+kh*16+8 .. +15

        const _Float16* bbase = (const _Float16*)(Bbuf + (ks & 1) * BSTEP_BYTES)
                                + (size_t)t0 * 512 + lane * 8;

        // Software-pipelined B-fragment reads: load tile i+1 while WMMA i runs,
        // via a 2-entry rotation buffer (constant indices after unroll).
        union BV { v16h v; v8h h[2]; } bb[2];
        bb[0].h[0] = *(const v8h*)(bbase);
        bb[0].h[1] = *(const v8h*)(bbase + 256);
#pragma unroll
        for (int i = 0; i < 12; ++i) {
            if (i + 1 < 12) {
                const _Float16* tb = bbase + (size_t)(i + 1) * 512;   // 1KB/tile
                bb[(i + 1) & 1].h[0] = *(const v8h*)(tb);
                bb[(i + 1) & 1].h[1] = *(const v8h*)(tb + 256);
            }
            acc[i] = __builtin_amdgcn_wmma_f32_16x16x32_f16(
                false, a.v, false, bb[i & 1].v, (short)0, acc[i], false, false);
        }
    }

    // ---- bias + LayerNorm stats (row = mrl + r + 8*kh) ----
    v8f s = {}, sq = {};
#pragma unroll
    for (int i = 0; i < 12; ++i) {
        acc[i] = acc[i] + bv[i];
        s  += acc[i];
        sq += acc[i] * acc[i];
    }
#pragma unroll
    for (int m = 1; m <= 8; m <<= 1) {
#pragma unroll
        for (int r = 0; r < 8; ++r) {
            s[r]  += __shfl_xor(s[r],  m, 32);
            sq[r] += __shfl_xor(sq[r], m, 32);
        }
    }
    if (c16 == 0) {
#pragma unroll
        for (int r = 0; r < 8; ++r) {
            int rl = mrl + r + 8 * kh;
            LN[rl * 4 + (w & 1) * 2 + 0] = s[r];
            LN[rl * 4 + (w & 1) * 2 + 1] = sq[r];
        }
    }
    __syncthreads();

    float mu[8], rstd[8];
#pragma unroll
    for (int r = 0; r < 8; ++r) {
        int rl = mrl + r + 8 * kh;
        float sm = LN[rl * 4 + 0] + LN[rl * 4 + 2];
        float qm = LN[rl * 4 + 1] + LN[rl * 4 + 3];
        float m  = sm * (1.0f / (float)DIMD);
        float v  = qm * (1.0f / (float)DIMD) - m * m;
        mu[r]   = m;
        rstd[r] = rsqrtf(v + 1e-5f);
    }

#pragma unroll
    for (int i = 0; i < 12; ++i) {
        int n = (t0 + i) * 16 + c16;
        float g  = gamma[n];
        float be = beta[n];
#pragma unroll
        for (int r = 0; r < 8; ++r) {
            int rl = mrl + r + 8 * kh;
            float val = (acc[i][r] - mu[r]) * rstd[r] * g + be;
            out[((size_t)(m0 + rl)) * DIMD + n] = val;
        }
    }
}

// ---------------------------------------------------------------------------
extern "C" void kernel_launch(void* const* d_in, const int* in_sizes, int n_in,
                              void* d_out, int out_size, void* d_ws, size_t ws_size,
                              hipStream_t stream) {
    const float* images   = (const float*)d_in[0];   // (64,3,224,224)
    const float* proj_w   = (const float*)d_in[1];   // (384,3,16,16) == (384,768)
    const float* proj_b   = (const float*)d_in[2];   // (384,)
    const float* ln_gamma = (const float*)d_in[3];   // (384,)
    const float* ln_beta  = (const float*)d_in[4];   // (384,)
    float* out = (float*)d_out;                      // 12544*384 x + 12544 entropy
    _Float16* wsB = (_Float16*)d_ws;                 // needs 576KB scratch

    // 1) pack weights into f16 LDS-image B layout (L2-resident afterwards)
    pack_w_kernel<<<(NT * KSTEPS * 32 + 255) / 256, 256, 0, stream>>>(proj_w, wsB);

    // 2) fused patch-embed GEMM + bias + LayerNorm + entropy
    vit_embed_kernel<<<MTOT / ROWS_WG, 256, SMEM_BYTES, stream>>>(
        images, proj_b, ln_gamma, ln_beta, wsB, out);
}
